// NetVLAD_21844203668356
// MI455X (gfx1250) — matrix operand order repaired and checked by reading, not compiled, not run on hardware
//
#include <hip/hip_runtime.h>
#include <hip/hip_bf16.h>

typedef __attribute__((ext_vector_type(16))) __bf16 v16bf;
typedef __attribute__((ext_vector_type(8)))  float  v8f;
typedef unsigned int u32x4 __attribute__((ext_vector_type(4)));
typedef int          i32x8 __attribute__((ext_vector_type(8)));
typedef int          i32x4 __attribute__((ext_vector_type(4)));

#define N_ 64
#define T_ 2048
#define C_ 256
#define K_ 64

union Frag16 { v16bf v; unsigned int u[8]; __bf16 h[16]; };
union Pack2  { __bf16 h[2]; unsigned int u; };

// ---------------------------------------------------------------------------
// TDM: issue a 2D tensor_load_to_lds (bf16 elements, data_size=2B).
// D# group0: count=1 | lds_addr | global_addr | type=2 ("image").
// D# group1: data_size=1(2B), tensor_dim0/1, tile_dim0/1, dim0 stride.
// Caller must be uniform (issued from one wave); completion via TENSORcnt.
// ---------------------------------------------------------------------------
__device__ __forceinline__ void tdm_load_2d(unsigned lds_off, const void* gaddr,
                                            unsigned dim0, unsigned dim1,
                                            unsigned tile0, unsigned tile1,
                                            unsigned stride0) {
    unsigned long long ga = (unsigned long long)gaddr;
    u32x4 g0;
    g0[0] = 1u;                                                  // count=1
    g0[1] = lds_off;                                             // lds_addr
    g0[2] = (unsigned)(ga & 0xffffffffu);                        // global lo
    g0[3] = (unsigned)((ga >> 32) & 0x01ffffffu) | 0x80000000u;  // hi | type=2
    i32x8 g1;
    g1[0] = (int)(1u << 16);                                     // data_size=2B
    g1[1] = (int)((dim0 & 0xffffu) << 16);                       // dim0 lo
    g1[2] = (int)((dim0 >> 16) | ((dim1 & 0xffffu) << 16));      // dim0 hi|dim1 lo
    g1[3] = (int)((dim1 >> 16) | (tile0 << 16));                 // dim1 hi|tile0
    g1[4] = (int)(tile1 & 0xffffu);                              // tile1, tile2=0
    g1[5] = (int)stride0;                                        // dim0 stride lo
    g1[6] = 0;
    g1[7] = 0;
    i32x4 z4 = {0, 0, 0, 0};
#if __clang_major__ >= 23
    i32x8 z8 = {0, 0, 0, 0, 0, 0, 0, 0};
    __builtin_amdgcn_tensor_load_to_lds(g0, g1, z4, z4, z8, 0);
#else
    __builtin_amdgcn_tensor_load_to_lds(g0, g1, z4, z4, 0);
#endif
}

__device__ __forceinline__ unsigned lds_off_of(const void* p) {
    // Generic pointers into LDS carry the LDS byte offset in the low 32 bits
    // (ISA 10.2 aperture model).
    return (unsigned)(size_t)p;
}

// ---------------------------------------------------------------------------
// Kernel 1: L2-normalize each x[n,t,:] row (C=256) and emit bf16 xb[n][t][c].
// ---------------------------------------------------------------------------
__global__ void nv_normalize(const float* __restrict__ x, __bf16* __restrict__ xb) {
    int w = threadIdx.x >> 5, L = threadIdx.x & 31;
    size_t row = (size_t)blockIdx.x * 8 + w;           // row = n*T + t
    const float4* s4 = (const float4*)(x + row * C_ + L * 8);
    float4 a = s4[0], b = s4[1];
    float v[8] = {a.x, a.y, a.z, a.w, b.x, b.y, b.z, b.w};
    float ss = 0.f;
    #pragma unroll
    for (int j = 0; j < 8; ++j) ss += v[j] * v[j];
    #pragma unroll
    for (int m = 1; m <= 16; m <<= 1) ss += __shfl_xor(ss, m, 32);
    float sc = 1.f / fmaxf(sqrtf(ss), 1e-12f);
    unsigned int* dst = (unsigned int*)(xb + row * C_ + L * 8);
    #pragma unroll
    for (int p = 0; p < 4; ++p) {
        Pack2 pk;
        pk.h[0] = (__bf16)(v[2 * p] * sc);
        pk.h[1] = (__bf16)(v[2 * p + 1] * sc);
        dst[p] = pk.u;
    }
}

// Convert W (f32 [K][C]) to bf16 once so TDM can stage it into LDS.
__global__ void nv_prep(const float* __restrict__ W, __bf16* __restrict__ Wbf) {
    int i = blockIdx.x * blockDim.x + threadIdx.x;
    Wbf[i] = (__bf16)W[i];
}

__global__ void nv_zero(float* __restrict__ p, int n) {
    int i = blockIdx.x * blockDim.x + threadIdx.x;
    if (i < n) p[i] = 0.f;
}

// ---------------------------------------------------------------------------
// Kernel 2: logits = xb @ W^T + b  (WMMA bf16, f32 acc), fused softmax over K,
// write aT[n][k][t] (bf16) and accumulate asum[n][k] (f32 atomics).
// W staged into LDS by the Tensor Data Mover.
// ---------------------------------------------------------------------------
__global__ void nv_assign(const __bf16* __restrict__ xb, const __bf16* __restrict__ Wbf,
                          const float* __restrict__ b, __bf16* __restrict__ aT,
                          float* __restrict__ asum) {
    __shared__ __bf16 Wl[K_ * C_];   // 32 KB, bf16 row-major [k][c]
    int tid = threadIdx.x;
    int w = tid >> 5, L = tid & 31, half = L >> 4, ln = L & 15;
    if (w == 0) {
        tdm_load_2d(lds_off_of(Wl), Wbf, K_ * C_, 1, K_ * C_, 1, K_ * C_);
        __builtin_amdgcn_s_wait_tensorcnt(0);
    }
    __syncthreads();

    int n  = blockIdx.x >> 4;
    int t0 = ((blockIdx.x & 15) * 8 + w) * 16;

    float bk[4];
    #pragma unroll
    for (int kt = 0; kt < 4; ++kt) bk[kt] = b[kt * 16 + ln];

    v8f d[4];
    #pragma unroll
    for (int kt = 0; kt < 4; ++kt)
        #pragma unroll
        for (int v = 0; v < 8; ++v) d[kt][v] = 0.f;

    const __bf16* xrow = xb + ((size_t)n * T_ + t0 + ln) * C_;
    for (int c0 = 0; c0 < C_; c0 += 32) {
        Frag16 af;   // A: 16x32 bf16, M = t row, K = c
        #pragma unroll
        for (int e2 = 0; e2 < 8; ++e2) {
            int kb = ((e2 >> 2) << 4) + (half << 3) + ((e2 & 3) << 1);
            af.u[e2] = *(const unsigned int*)(xrow + c0 + kb);
        }
        #pragma unroll
        for (int kt = 0; kt < 4; ++kt) {
            Frag16 bf;   // B: 32x16 bf16, K = c, N = k
            #pragma unroll
            for (int e2 = 0; e2 < 8; ++e2) {
                int c = c0 + ((e2 >> 2) << 4) + (half << 3) + ((e2 & 3) << 1);
                bf.u[e2] = *(const unsigned int*)(&Wl[(kt * 16 + ln) * C_ + c]);
            }
            d[kt] = __builtin_amdgcn_wmma_f32_16x16x32_bf16(
                false, af.v, false, bf.v, (short)0, d[kt], false, false);
        }
    }

    // Fused softmax over K=64 (4 tiles x 16 lanes within this half).
    float asumAcc[4] = {0.f, 0.f, 0.f, 0.f};
    #pragma unroll
    for (int v = 0; v < 8; ++v) {
        float l0 = d[0][v] + bk[0], l1 = d[1][v] + bk[1];
        float l2 = d[2][v] + bk[2], l3 = d[3][v] + bk[3];
        float mx = fmaxf(fmaxf(l0, l1), fmaxf(l2, l3));
        #pragma unroll
        for (int m = 1; m <= 8; m <<= 1) mx = fmaxf(mx, __shfl_xor(mx, m, 32));
        float e0 = __expf(l0 - mx), e1 = __expf(l1 - mx);
        float e2 = __expf(l2 - mx), e3 = __expf(l3 - mx);
        float s = e0 + e1 + e2 + e3;
        #pragma unroll
        for (int m = 1; m <= 8; m <<= 1) s += __shfl_xor(s, m, 32);
        float inv = 1.f / s;
        e0 *= inv; e1 *= inv; e2 *= inv; e3 *= inv;
        asumAcc[0] += e0; asumAcc[1] += e1; asumAcc[2] += e2; asumAcc[3] += e3;
        int t = t0 + v + (half << 3);
        aT[((size_t)n * K_ +  0 + ln) * T_ + t] = (__bf16)e0;
        aT[((size_t)n * K_ + 16 + ln) * T_ + t] = (__bf16)e1;
        aT[((size_t)n * K_ + 32 + ln) * T_ + t] = (__bf16)e2;
        aT[((size_t)n * K_ + 48 + ln) * T_ + t] = (__bf16)e3;
    }
    #pragma unroll
    for (int kt = 0; kt < 4; ++kt)
        atomicAdd(&asum[n * K_ + kt * 16 + ln], asumAcc[kt]);
}

// ---------------------------------------------------------------------------
// Kernel 3: ax = a^T @ xb per n (WMMA, contraction over t), then fused
// vlad = ax - asum*centroids, intra-L2-norm over C, exact global L2 norm.
// One block per n; wave w owns ALL 4 k-tiles x c-tiles {2w, 2w+1} so every
// B fragment is loaded exactly once per wave. x tiles arrive via TDM with
// double buffering (TENSORcnt) so the DMA overlaps the WMMAs.
// ---------------------------------------------------------------------------
__global__ void nv_vlad(const __bf16* __restrict__ xb, const __bf16* __restrict__ aT,
                        const float* __restrict__ asum, const float* __restrict__ cent,
                        float* __restrict__ out) {
    __shared__ __bf16 xstage[2][32 * C_];   // 2 x 16 KB x-tile double buffer
    __shared__ float  partial[8 * K_];      // per-wave row-k sumsq partials
    __shared__ float  rnorm_s[K_];
    __shared__ float  gred[K_];
    __shared__ float  grn_s;

    int tid = threadIdx.x, w = tid >> 5, L = tid & 31, half = L >> 4, ln = L & 15;
    int n = blockIdx.x;

    v8f acc[8];   // acc[kt*2+j]: k-tile kt (0..3), c-tile 2w+j (j=0,1)
    #pragma unroll
    for (int i = 0; i < 8; ++i)
        #pragma unroll
        for (int v = 0; v < 8; ++v) acc[i][v] = 0.f;

    unsigned lds0 = lds_off_of(&xstage[0][0]);
    unsigned lds1 = lds_off_of(&xstage[1][0]);
    if (w == 0)   // preload first tile
        tdm_load_2d(lds0, xb + (size_t)n * T_ * C_, C_, 32, C_, 32, C_);

    for (int t0 = 0; t0 < T_; t0 += 32) {
        int p = (t0 >> 5) & 1;
        if (w == 0) {
            if (t0 + 32 < T_) {
                tdm_load_2d(p ? lds0 : lds1,
                            xb + ((size_t)n * T_ + t0 + 32) * C_, C_, 32, C_, 32, C_);
                __builtin_amdgcn_s_wait_tensorcnt(1);
            } else {
                __builtin_amdgcn_s_wait_tensorcnt(0);
            }
        }
        __syncthreads();   // tile p visible to all waves

        const __bf16* xt = xstage[p];
        Frag16 bf[2];      // B: K = t (rows), N = c -> transposed gather
        #pragma unroll
        for (int j = 0; j < 2; ++j) {
            int c = (2 * w + j) * 16 + ln;
            #pragma unroll
            for (int e = 0; e < 16; ++e)
                bf[j].h[e] = xt[(((e >> 3) << 4) + (half << 3) + (e & 7)) * C_ + c];
        }
        #pragma unroll
        for (int kt = 0; kt < 4; ++kt) {
            Frag16 af;     // A = a^T tile: M = k, K = t; aT is t-contiguous
            const __bf16* arow = aT + ((size_t)n * K_ + kt * 16 + ln) * T_ + t0;
            #pragma unroll
            for (int e2 = 0; e2 < 8; ++e2) {
                int kb = ((e2 >> 2) << 4) + (half << 3) + ((e2 & 3) << 1);
                af.u[e2] = *(const unsigned int*)(arow + kb);
            }
            acc[kt * 2 + 0] = __builtin_amdgcn_wmma_f32_16x16x32_bf16(
                false, af.v, false, bf[0].v, (short)0, acc[kt * 2 + 0], false, false);
            acc[kt * 2 + 1] = __builtin_amdgcn_wmma_f32_16x16x32_bf16(
                false, af.v, false, bf[1].v, (short)0, acc[kt * 2 + 1], false, false);
        }
        __syncthreads();   // tile p fully consumed before its buffer is reloaded
    }

    // vlad = ax - asum[k]*centroids[k][c]; then row sum-of-squares partials.
    #pragma unroll
    for (int kt = 0; kt < 4; ++kt) {
        float av[8];
        #pragma unroll
        for (int v = 0; v < 8; ++v) av[v] = asum[n * K_ + kt * 16 + v + (half << 3)];
        #pragma unroll
        for (int j = 0; j < 2; ++j) {
            int c = (2 * w + j) * 16 + ln;
            #pragma unroll
            for (int v = 0; v < 8; ++v) {
                int k = kt * 16 + v + (half << 3);
                acc[kt * 2 + j][v] -= av[v] * cent[k * C_ + c];
            }
        }
        #pragma unroll
        for (int v = 0; v < 8; ++v) {
            float p = acc[kt * 2 + 0][v] * acc[kt * 2 + 0][v]
                    + acc[kt * 2 + 1][v] * acc[kt * 2 + 1][v];
            #pragma unroll
            for (int m = 1; m <= 8; m <<= 1) p += __shfl_xor(p, m, 32);
            if (ln == 0) partial[w * K_ + kt * 16 + (half << 3) + v] = p;
        }
    }
    __syncthreads();
    if (tid < K_) {
        float tot = 0.f;
        #pragma unroll
        for (int ww = 0; ww < 8; ++ww) tot += partial[ww * K_ + tid];
        float rn = 1.f / fmaxf(sqrtf(tot), 1e-12f);
        rnorm_s[tid] = rn;
        gred[tid] = tot * rn * rn;   // exact contribution to global sumsq
    }
    __syncthreads();
    if (tid == 0) {
        float g = 0.f;
        for (int k = 0; k < K_; ++k) g += gred[k];
        grn_s = 1.f / fmaxf(sqrtf(g), 1e-12f);
    }
    __syncthreads();
    float grn = grn_s;
    #pragma unroll
    for (int kt = 0; kt < 4; ++kt)
        #pragma unroll
        for (int v = 0; v < 8; ++v) {
            int k = kt * 16 + v + (half << 3);
            float sc = rnorm_s[k] * grn;
            #pragma unroll
            for (int j = 0; j < 2; ++j) {
                int c = (2 * w + j) * 16 + ln;
                out[(size_t)n * (K_ * C_) + k * C_ + c] = acc[kt * 2 + j][v] * sc;
            }
        }
}

// ---------------------------------------------------------------------------
extern "C" void kernel_launch(void* const* d_in, const int* in_sizes, int n_in,
                              void* d_out, int out_size, void* d_ws, size_t ws_size,
                              hipStream_t stream) {
    (void)in_sizes; (void)n_in; (void)out_size; (void)ws_size;
    const float* x    = (const float*)d_in[0];
    const float* W    = (const float*)d_in[1];
    const float* b    = (const float*)d_in[2];
    const float* cent = (const float*)d_in[3];
    float* out = (float*)d_out;

    char* ws = (char*)d_ws;
    size_t off = 0;
    __bf16* xb  = (__bf16*)(ws + off); off += (size_t)N_ * T_ * C_ * sizeof(__bf16); // 64 MB
    __bf16* aT  = (__bf16*)(ws + off); off += (size_t)N_ * K_ * T_ * sizeof(__bf16); // 16 MB
    float*  asum = (float*)(ws + off); off += (size_t)N_ * K_ * sizeof(float);       // 16 KB
    __bf16* Wbf = (__bf16*)(ws + off);                                               // 32 KB

    nv_normalize<<<(N_ * T_) / 8, 256, 0, stream>>>(x, xb);
    nv_prep<<<(K_ * C_) / 256, 256, 0, stream>>>(W, Wbf);
    nv_zero<<<(N_ * K_ + 255) / 256, 256, 0, stream>>>(asum, N_ * K_);
    nv_assign<<<N_ * (T_ / 16) / 8, 256, 0, stream>>>(xb, Wbf, b, aT, asum);
    nv_vlad<<<N_, 256, 0, stream>>>(xb, aT, asum, cent, out);
}
